// ContinuousFilterConvolution_9560597201471
// MI455X (gfx1250) — matrix operand order repaired and checked by reading, not compiled, not run on hardware
//
#include <hip/hip_runtime.h>
#include <hip/hip_bf16.h>
#include <stdint.h>

typedef __attribute__((ext_vector_type(16))) __bf16       v16bf;
typedef __attribute__((ext_vector_type(2)))  __bf16       v2bf;
typedef __attribute__((ext_vector_type(8)))  float        v8f;
typedef __attribute__((ext_vector_type(2)))  float        v2f;
typedef __attribute__((ext_vector_type(4)))  unsigned int v4u;
typedef __attribute__((ext_vector_type(2)))  unsigned int v2u;

#define HIDDEN 256
#define NBASES 128
#define TM 16              // edges per tile (WMMA M)
#define WAVES 16           // one wave per 16-col N-tile of HIDDEN
#define THREADS (WAVES * 32)

union Frag16 {
    v16bf v;
    unsigned short u[16];
    v4u q[2];
};

// single-instruction relu: v_med3_f32(x, 0, +big)
__device__ __forceinline__ float relu(float x) {
    return __builtin_amdgcn_fmed3f(x, 0.0f, 3.4e38f);
}

// packed f32x2 -> bf16x2: vector convert lowers to one v_cvt_pk_bf16_f32
__device__ __forceinline__ unsigned int pk2bf(float a, float b) {
    v2f f = {a, b};
    v2bf p = __builtin_convertvector(f, v2bf);
    return __builtin_bit_cast(unsigned int, p);
}

// ---- prep: W1 [128x256] -> W1t bf16 [256x128]; W2 [256x256] -> W2t bf16 [256x256]
__global__ void cfconv_prep(const float* __restrict__ W1, const float* __restrict__ W2,
                            __bf16* __restrict__ W1t, __bf16* __restrict__ W2t) {
    int i = blockIdx.x * blockDim.x + threadIdx.x;   // 0 .. 65535
    if (i < NBASES * HIDDEN) {
        int k = i / HIDDEN, n = i % HIDDEN;
        W1t[n * NBASES + k] = (__bf16)W1[i];
    }
    if (i < HIDDEN * HIDDEN) {
        int k = i / HIDDEN, n = i % HIDDEN;
        W2t[n * HIDDEN + k] = (__bf16)W2[i];
    }
}

__global__ void cfconv_zero(float* __restrict__ out, int n4) {
    int i = blockIdx.x * blockDim.x + threadIdx.x;
    if (i < n4) {
        v4u z = {0u, 0u, 0u, 0u};
        reinterpret_cast<v4u*>(out)[i] = z;
    }
}

__global__ __launch_bounds__(THREADS)
void cfconv_main(const float* __restrict__ X, const float* __restrict__ R,
                 const __bf16* __restrict__ W1t,
                 const __bf16* __restrict__ W2t,
                 const float* __restrict__ mu,
                 const int* __restrict__ src, const int* __restrict__ dst,
                 float* __restrict__ H, int ntiles) {
    __shared__ __align__(16) unsigned short rbf_lds[TM * NBASES];  // 4 KB
    __shared__ __align__(16) __bf16 h_lds[TM * HIDDEN];            // 8 KB
    __shared__ float D_lds[TM];
    __shared__ int   s_lds[TM];
    __shared__ int   d_lds[TM];

    const int tid  = threadIdx.x;
    const int lane = tid & 31;
    const int wave = tid >> 5;            // N-tile 0..15
    const int nloc = lane & 15;
    const int half = lane >> 4;           // 0 or 1
    const int col  = wave * 16 + nloc;    // global hidden column for B/C/D
    const int arow = lane & 15;           // A-fragment row

    const float mu0   = mu[0];
    const float step  = mu[1] - mu[0];
    const float gamma = 1.0f / (step * step);

    // Persistent B fragments (32x16 bf16 B layout: lanes 0-15 K=0..7,16..23,
    // lanes 16-31 K=8..15,24..31; matches A fragment order).
    Frag16 b1[4];
#pragma unroll
    for (int kb = 0; kb < 4; ++kb) {
        const v4u* p = reinterpret_cast<const v4u*>(W1t + col * NBASES + kb * 32 + half * 8);
        b1[kb].q[0] = p[0];
        b1[kb].q[1] = p[2];
    }
    Frag16 b2[8];
#pragma unroll
    for (int kb = 0; kb < 8; ++kb) {
        const v4u* p = reinterpret_cast<const v4u*>(W2t + col * HIDDEN + kb * 32 + half * 8);
        b2[kb].q[0] = p[0];
        b2[kb].q[1] = p[2];
    }

    for (int tile = blockIdx.x; tile < ntiles; tile += gridDim.x) {
        // ---- Phase 0: squared distances + edge endpoints for 16 edges
        if (tid < TM) {
            int e = tile * TM + tid;
            int s = src[e];
            int d = dst[e];
            float dx = R[3 * s + 0] - R[3 * d + 0];
            float dy = R[3 * s + 1] - R[3 * d + 1];
            float dz = R[3 * s + 2] - R[3 * d + 2];
            s_lds[tid] = s;
            d_lds[tid] = d;
            D_lds[tid] = dx * dx + dy * dy + dz * dz;
        }
        __syncthreads();

        // ---- Phase 0b: RBF tile [16 x 128] bf16 into LDS (4 values/thread)
        {
            int base = tid * 4;            // 512*4 == TM*NBASES
            int row  = base >> 7;
            int k0   = base & 127;
            float D  = D_lds[row];
            float t0 = D - (mu0 + (float)(k0 + 0) * step);
            float t1 = D - (mu0 + (float)(k0 + 1) * step);
            float t2 = D - (mu0 + (float)(k0 + 2) * step);
            float t3 = D - (mu0 + (float)(k0 + 3) * step);
            float e0 = __expf(-gamma * t0 * t0);
            float e1 = __expf(-gamma * t1 * t1);
            float e2 = __expf(-gamma * t2 * t2);
            float e3 = __expf(-gamma * t3 * t3);
            v2u pk = {pk2bf(e0, e1), pk2bf(e2, e3)};
            *reinterpret_cast<v2u*>(&rbf_lds[base]) = pk;
        }
        __syncthreads();

        // ---- Phase 1: h = relu(rbf @ W1), K = 128 in 4 steps of 32
        v8f acc1;
#pragma unroll
        for (int i = 0; i < 8; ++i) acc1[i] = 0.0f;
        {
            Frag16 a_cur, a_nxt;
            const unsigned short* ap0 = rbf_lds + arow * NBASES + half * 8;
            a_cur.q[0] = *reinterpret_cast<const v4u*>(ap0);        // K khalf..khalf+7
            a_cur.q[1] = *reinterpret_cast<const v4u*>(ap0 + 16);   // K 16+khalf..
#pragma unroll
            for (int kb = 0; kb < 4; ++kb) {
                if (kb < 3) {
                    const unsigned short* ap = rbf_lds + arow * NBASES + (kb + 1) * 32 + half * 8;
                    a_nxt.q[0] = *reinterpret_cast<const v4u*>(ap);
                    a_nxt.q[1] = *reinterpret_cast<const v4u*>(ap + 16);
                }
                acc1 = __builtin_amdgcn_wmma_f32_16x16x32_bf16(
                    false, a_cur.v, false, b1[kb].v, (short)0, acc1, false, false);
                a_cur = a_nxt;
            }
        }
        // relu -> bf16 -> LDS h tile [16 x 256]
#pragma unroll
        for (int v = 0; v < 8; ++v) {
            int row = v + half * 8;                    // C/D layout
            h_lds[row * HIDDEN + col] = (__bf16)relu(acc1[v]);
        }
        __syncthreads();

        // ---- Phase 2: M = relu(h @ W2), K = 256 in 8 steps of 32
        v8f acc2;
#pragma unroll
        for (int i = 0; i < 8; ++i) acc2[i] = 0.0f;
        {
            Frag16 a_cur, a_nxt;
            const __bf16* ap0 = h_lds + arow * HIDDEN + half * 8;
            a_cur.q[0] = *reinterpret_cast<const v4u*>(ap0);
            a_cur.q[1] = *reinterpret_cast<const v4u*>(ap0 + 16);
#pragma unroll
            for (int kb = 0; kb < 8; ++kb) {
                if (kb < 7) {
                    const __bf16* ap = h_lds + arow * HIDDEN + (kb + 1) * 32 + half * 8;
                    a_nxt.q[0] = *reinterpret_cast<const v4u*>(ap);
                    a_nxt.q[1] = *reinterpret_cast<const v4u*>(ap + 16);
                }
                acc2 = __builtin_amdgcn_wmma_f32_16x16x32_bf16(
                    false, a_cur.v, false, b2[kb].v, (short)0, acc2, false, false);
                a_cur = a_nxt;
            }
        }

        // ---- Phase 3: msg = X[src] * relu(M); scatter-add into H[dest]
#pragma unroll
        for (int v = 0; v < 8; ++v) {
            int row = v + half * 8;
            int s = s_lds[row];
            int d = d_lds[row];
            float Mv = relu(acc2[v]);
            float x = X[s * HIDDEN + col];
            __hip_atomic_fetch_add(&H[d * HIDDEN + col], x * Mv,
                                   __ATOMIC_RELAXED, __HIP_MEMORY_SCOPE_AGENT);
        }
        __syncthreads();   // LDS reused next tile
    }
}

extern "C" void kernel_launch(void* const* d_in, const int* in_sizes, int n_in,
                              void* d_out, int out_size, void* d_ws, size_t ws_size,
                              hipStream_t stream) {
    const float* X  = (const float*)d_in[0];
    const float* R  = (const float*)d_in[1];
    const float* W1 = (const float*)d_in[2];
    const float* W2 = (const float*)d_in[3];
    const float* mu = (const float*)d_in[4];
    const int*  src = (const int*)d_in[5];
    const int*  dst = (const int*)d_in[6];
    float* H = (float*)d_out;

    __bf16* W1t = (__bf16*)d_ws;                         // 64 KB
    __bf16* W2t = W1t + NBASES * HIDDEN;                 // 128 KB

    // 1) bf16 transposed weights
    cfconv_prep<<<(HIDDEN * HIDDEN + 255) / 256, 256, 0, stream>>>(W1, W2, W1t, W2t);

    // 2) zero output accumulator (d_out is poisoned)
    int n4 = out_size / 4;
    cfconv_zero<<<(n4 + 255) / 256, 256, 0, stream>>>(H, n4);

    // 3) main fused kernel
    int E = in_sizes[5];
    int ntiles = E / TM;
    int grid = 2048;
    if (grid > ntiles) grid = ntiles;
    cfconv_main<<<grid, THREADS, 0, stream>>>(X, R, W1t, W2t, mu, src, dst, H, ntiles);
}
